// ColorCC_317827580560
// MI455X (gfx1250) — matrix operand order repaired and checked by reading, not compiled, not run on hardware
//
#include <hip/hip_runtime.h>
#include <hip/hip_bf16.h>

typedef __attribute__((ext_vector_type(16))) _Float16 v16h;
typedef __attribute__((ext_vector_type(8)))  _Float16 v8h;
typedef __attribute__((ext_vector_type(8)))  float    v8f;

// ---- layer tables (block order = sorted dict keys: fc11,fc21,fc22,fc23,fc31,fc41,fc_end) ----
// layer l = block*3 + sublayer
constexpr int L_DIN[21]  = {9,18,18,  18,54,54,  9,54,54,  486,54,54,  54,18,18,  18,6,6,  102,51,51};
constexpr int L_DOUT[21] = {18,18,18, 54,54,54,  54,54,54, 54,54,54,   18,18,18,  6,6,6,   51,51,51};
constexpr int L_KT[21]   = {1,1,1,    1,2,2,     1,2,2,    16,2,2,     2,1,1,     1,1,1,   4,2,2};
constexpr int L_NT[21]   = {2,2,2,    4,4,4,     4,4,4,    4,4,4,      2,2,2,     1,1,1,   4,4,4};
// cumulative offsets: B-panel halfs (tile = KT*NT*512), bias floats (NT*16)
constexpr int BOFF[21]  = {0,1024,2048,3072,5120,9216,13312,15360,19456,23552,56320,
                           60416,64512,66560,67584,68608,69120,69632,70144,78336,82432};
constexpr int BTOT      = 86528;
constexpr int BIOFF[21] = {0,32,64,96,160,224,288,352,416,480,544,608,672,704,736,768,784,800,816,880,944};
constexpr int BITOT     = 1008;

struct WPtrs { const float* W[21]; const float* b[21]; };

// ---------------- weight/bias packing ----------------
// B-panel layout: for layer l, tile t = kt*NT + nt, element (lane, i):
//   n = nt*16 + (lane & 15);  k = kt*32 + (lane>>4)*8 + (i&7) + (i>>3)*16   (B: ISA table order, matches A)
__global__ __launch_bounds__(256) void k0_pack(WPtrs p, _Float16* __restrict__ Bp,
                                               float* __restrict__ Bias) {
    int idx = blockIdx.x * 256 + threadIdx.x;
    if (idx < BTOT) {
        int l = 0;
        while (l < 20 && idx >= BOFF[l + 1]) ++l;
        int rem  = idx - BOFF[l];
        int t    = rem >> 9, w = rem & 511;
        int lane = w >> 4,  i = w & 15;
        int nt   = t % L_NT[l], kt = t / L_NT[l];
        int k    = kt * 32 + ((lane >> 4) << 3) + (i & 7) + ((i >> 3) << 4);
        int n    = nt * 16 + (lane & 15);
        float v  = 0.f;
        if (k < L_DIN[l] && n < L_DOUT[l]) v = p.W[l][n * L_DIN[l] + k];
        Bp[idx] = (_Float16)v;
    } else {
        int bi = idx - BTOT;
        if (bi < BITOT) {
            int l = 0;
            while (l < 20 && bi >= BIOFF[l + 1]) ++l;
            int n = bi - BIOFF[l];
            Bias[bi] = (n < L_DOUT[l]) ? p.b[l][n] : 0.f;
        }
    }
}

// ---------------- wave-level WMMA GEMM (layer templated; kt-outer, NT accumulators) ----------
// OUT[16 x NT*16] = act( A[16 x KT*32] (f16, LDS) * Bpanel + bias )
// A fragment per ISA 16-bit 16x32 layout: row = lane&15, kbase = (lane>>4)*8,
//   halfs 0..7 -> K = kbase+0..7, halfs 8..15 -> K = kbase+16..23 (two 16B LDS loads).
// C/D layout: n = lane&15, m = r + 8*(lane>>4).
template <int L, bool RELU, typename Writer>
__device__ __forceinline__ void wave_gemm(const _Float16* __restrict__ A, int strideA,
                                          const _Float16* __restrict__ Bp,
                                          const float* __restrict__ Bias, Writer&& wr) {
    constexpr int KT = L_KT[L], NT = L_NT[L];
    const int lane = threadIdx.x & 31;
    const int row  = lane & 15;
    const int kb   = (lane >> 4) * 8;
    const int mb   = (lane >> 4) * 8;
    const int n    = lane & 15;
    const _Float16* bp = Bp + BOFF[L] + lane * 16;
    v8f acc[NT];
#pragma unroll
    for (int t = 0; t < NT; ++t) acc[t] = v8f{};
#pragma unroll
    for (int kt = 0; kt < KT; ++kt) {
        const _Float16* ap = A + row * strideA + kt * 32 + kb;
        v8h lo = *(const v8h*)ap;
        v8h hi = *(const v8h*)(ap + 16);
        v16h a;
#pragma unroll
        for (int i = 0; i < 8; ++i) { a[i] = lo[i]; a[i + 8] = hi[i]; }
#pragma unroll
        for (int nt = 0; nt < NT; ++nt) {
            v16h b = *(const v16h*)(bp + (size_t)(kt * NT + nt) * 512);
            acc[nt] = __builtin_amdgcn_wmma_f32_16x16x32_f16(false, a, false, b, (short)0,
                                                             acc[nt], false, false);
        }
    }
#pragma unroll
    for (int nt = 0; nt < NT; ++nt) {
        float bv = Bias[BIOFF[L] + nt * 16 + n];
#pragma unroll
        for (int r = 0; r < 8; ++r) {
            float v = acc[nt][r] + bv;
            if (RELU) v = fmaxf(v, 0.f);
            wr(nt * 16 + n, mb + r, v);
        }
    }
}

// GEMM to LDS with zero-fill of columns [NT*16, PAD)
template <int L, bool RELU, int PAD>
__device__ __forceinline__ void layer_lds(const _Float16* A, int sA, _Float16* O, int sO,
                                          const _Float16* Bp, const float* Bias) {
    wave_gemm<L, RELU>(A, sA, Bp, Bias,
                       [&](int col, int m, float v) { O[m * sO + col] = (_Float16)v; });
    constexpr int nc = L_NT[L] * 16;
    if constexpr (PAD > nc) {
        const int lane = threadIdx.x & 31;
#pragma unroll
        for (int idx0 = 0; idx0 < (PAD - nc) * 16; idx0 += 32) {
            int idx = idx0 + lane;
            O[(idx & 15) * sO + nc + (idx >> 4)] = (_Float16)0.f;
        }
    }
}

// ---------------- kernel 1: per-point chain fc11->fc21->fc31->fc41 ----------------
__global__ __launch_bounds__(256)
void k1_pointchain(const float* __restrict__ fea, const _Float16* __restrict__ Bp,
                   const float* __restrict__ Bias, _Float16* __restrict__ fea2h,
                   _Float16* __restrict__ cat, int ntiles) {
    __shared__ __attribute__((aligned(32))) _Float16 sbuf[8][2][16][64];  // 32 KB
    const int wave = threadIdx.x >> 5, lane = threadIdx.x & 31;
    const int tile = blockIdx.x * 8 + wave;
    if (tile >= ntiles) return;
    const int base = tile * 16;
    _Float16* b0 = &sbuf[wave][0][0][0];
    _Float16* b1 = &sbuf[wave][1][0][0];

    // stage fea tile -> b0[16][0..31] (f16, zero-padded)
    for (int idx = lane; idx < 512; idx += 32) {
        int r = idx >> 5, c = idx & 31;
        float v = (c < 9) ? fea[(size_t)(base + r) * 9 + c] : 0.f;
        b0[r * 64 + c] = (_Float16)v;
    }
    // cat[:, 96:102] = fea[:, :6]
    for (int idx = lane; idx < 96; idx += 32) {
        int r = idx / 6, c = idx - r * 6;
        cat[(size_t)(base + r) * 102 + 96 + c] = (_Float16)fea[(size_t)(base + r) * 9 + c];
    }

    // fc11 (9->18->18->18, relu at end) -> fea1 to cat[:,78:96]
    layer_lds<0, false, 32>(b0, 64, b1, 64, Bp, Bias);
    layer_lds<1, false, 32>(b1, 64, b0, 64, Bp, Bias);
    wave_gemm<2, true>(b0, 64, Bp, Bias, [&](int col, int m, float v) {
        b1[m * 64 + col] = (_Float16)v;
        if (col < 18) cat[(size_t)(base + m) * 102 + 78 + col] = (_Float16)v;
    });
    // fc21 (18->54->54->54) -> fea2 to gather table (f16, stride 64)
    layer_lds<3, false, 64>(b1, 64, b0, 64, Bp, Bias);
    layer_lds<4, false, 64>(b0, 64, b1, 64, Bp, Bias);
    wave_gemm<5, true>(b1, 64, Bp, Bias, [&](int col, int m, float v) {
        b0[m * 64 + col] = (_Float16)v;
        if (col < 54) fea2h[(size_t)(base + m) * 64 + col] = (_Float16)v;
    });
    // fc31 (54->18->18->18) -> fea3 to cat[:,6:24]
    layer_lds<12, false, 32>(b0, 64, b1, 64, Bp, Bias);
    layer_lds<13, false, 32>(b1, 64, b0, 64, Bp, Bias);
    wave_gemm<14, true>(b0, 64, Bp, Bias, [&](int col, int m, float v) {
        b1[m * 64 + col] = (_Float16)v;
        if (col < 18) cat[(size_t)(base + m) * 102 + 6 + col] = (_Float16)v;
    });
    // fc41 (18->6->6->6) -> fea4 to cat[:,0:6]
    layer_lds<15, false, 32>(b1, 64, b0, 64, Bp, Bias);
    layer_lds<16, false, 32>(b0, 64, b1, 64, Bp, Bias);
    wave_gemm<17, true>(b1, 64, Bp, Bias, [&](int col, int m, float v) {
        if (col < 6) cat[(size_t)(base + m) * 102 + col] = (_Float16)v;
    });
}

// ---------------- kernel 2: neighbor fc22 chain, product, fc23 ----------------
__global__ __launch_bounds__(64)
void k2_neighbors(const float* __restrict__ fea, const long long* __restrict__ link,
                  const _Float16* __restrict__ Bp, const float* __restrict__ Bias,
                  const _Float16* __restrict__ fea2h, _Float16* __restrict__ cat,
                  int ntiles) {
    __shared__ __attribute__((aligned(32))) _Float16 big[2][16][512];   // 32 KB
    __shared__ __attribute__((aligned(32))) _Float16 pb[2][2][16][64];  // 8 KB
    __shared__ int   slink[2][16][9];
    __shared__ float sfea[2][16][9];
    const int wave = threadIdx.x >> 5, lane = threadIdx.x & 31;
    const int tile = blockIdx.x * 2 + wave;
    if (tile >= ntiles) return;
    const int base = tile * 16;
    _Float16* b0 = &pb[wave][0][0][0];
    _Float16* b1 = &pb[wave][1][0][0];
    _Float16* bg = &big[wave][0][0];
    int   (*lnk)[9] = slink[wave];
    float (*own)[9] = sfea[wave];

    // cache this tile's link table and own fea rows (16x9) in LDS
    for (int idx = lane; idx < 144; idx += 32) {
        int r = idx / 9, c = idx - r * 9;
        lnk[r][c] = (int)link[(size_t)(base + r) * 9 + c];
        own[r][c] = fea[(size_t)(base + r) * 9 + c];
    }

    union H2 { unsigned u; _Float16 h[2]; };
    const int cpair = lane * 2;  // paired-f16 column for gather/product

    for (int j = 0; j < 9; ++j) {
        // ninei[:, j, :] = fea[link] - fea  -> b0[16][0..31]
        for (int idx = lane; idx < 512; idx += 32) {
            int r = idx >> 5, c = idx & 31;
            float v = 0.f;
            if (c < 9) v = fea[(size_t)lnk[r][j] * 9 + c] - own[r][c];
            b0[r * 64 + c] = (_Float16)v;
        }
        // fc22 chain (9->54->54->54, relu at end)
        layer_lds<6, false, 64>(b0, 64, b1, 64, Bp, Bias);
        layer_lds<7, false, 64>(b1, 64, b0, 64, Bp, Bias);
        layer_lds<8, true, 64>(b0, 64, b1, 64, Bp, Bias);
        // fea2[link] * fc22(ninei) -> big[:, j*54 .. j*54+53]   (paired f16, b32 ops)
        if (cpair < 54) {
#pragma unroll 4
            for (int r = 0; r < 16; ++r) {
                int nb = lnk[r][j];
                H2 g, x, o;
                g.u = *(const unsigned*)(fea2h + (size_t)nb * 64 + cpair);
                x.u = *(const unsigned*)(b1 + r * 64 + cpair);
                o.h[0] = g.h[0] * x.h[0];
                o.h[1] = g.h[1] * x.h[1];
                *(unsigned*)(bg + r * 512 + j * 54 + cpair) = o.u;
            }
        }
    }
    // zero pad cols 486..511
    if (lane < 26) {
#pragma unroll
        for (int r = 0; r < 16; ++r) bg[r * 512 + 486 + lane] = (_Float16)0.f;
    }
    // fc23 (486->54->54->54, relu) -> cat[:,24:78]
    layer_lds<9, false, 64>(bg, 512, b0, 64, Bp, Bias);
    layer_lds<10, false, 64>(b0, 64, b1, 64, Bp, Bias);
    wave_gemm<11, true>(b1, 64, Bp, Bias, [&](int col, int m, float v) {
        if (col < 54) cat[(size_t)(base + m) * 102 + 24 + col] = (_Float16)v;
    });
}

// ---------------- kernel 3: fc_end head on cat (src=0) and cfe (src=1) ----------------
__global__ __launch_bounds__(128)
void k3_head(const _Float16* __restrict__ cat, const float* __restrict__ cfe,
             const _Float16* __restrict__ Bp, const float* __restrict__ Bias,
             float* __restrict__ out, int ntiles) {
    __shared__ __attribute__((aligned(32))) _Float16 a0[4][16][128];    // 16 KB
    __shared__ __attribute__((aligned(32))) _Float16 pb[4][2][16][64];  // 16 KB
    const int wave = threadIdx.x >> 5, lane = threadIdx.x & 31;
    const int gw = blockIdx.x * 4 + wave;
    const int tile = gw >> 1, src = gw & 1;
    if (tile >= ntiles) return;
    const int base = tile * 16;
    _Float16* A  = &a0[wave][0][0];
    _Float16* b0 = &pb[wave][0][0][0];
    _Float16* b1 = &pb[wave][1][0][0];

    for (int idx = lane; idx < 2048; idx += 32) {  // 16*128
        int r = idx >> 7, c = idx & 127;
        float v = 0.f;
        if (c < 102)
            v = (src == 0) ? (float)cat[(size_t)(base + r) * 102 + c]
                           : cfe[(size_t)(base + r) * 102 + c];
        A[r * 128 + c] = (_Float16)v;
    }
    layer_lds<18, false, 64>(A, 128, b0, 64, Bp, Bias);
    layer_lds<19, false, 64>(b0, 64, b1, 64, Bp, Bias);
    wave_gemm<20, true>(b1, 64, Bp, Bias, [&](int col, int m, float v) {
        if (col < 51) out[(size_t)(base + m) * 102 + src * 51 + col] = v;
    });
}

extern "C" void kernel_launch(void* const* d_in, const int* in_sizes, int n_in,
                              void* d_out, int out_size, void* d_ws, size_t ws_size,
                              hipStream_t stream) {
    const float*     fea  = (const float*)d_in[0];
    const long long* link = (const long long*)d_in[1];
    const float*     cfe  = (const float*)d_in[2];

    WPtrs wp;
    for (int blk = 0; blk < 7; ++blk)
        for (int li = 0; li < 3; ++li) {
            wp.W[blk * 3 + li] = (const float*)d_in[3 + blk * 6 + li * 2];
            wp.b[blk * 3 + li] = (const float*)d_in[3 + blk * 6 + li * 2 + 1];
        }

    const int N = in_sizes[0] / 9;       // 300000
    const int ntiles = (N + 15) / 16;    // 18750 (exact)

    // workspace layout (bytes)
    char* ws = (char*)d_ws;
    _Float16* Bp    = (_Float16*)(ws + 0);                            // 86528 halfs
    float*    Bias  = (float*)(ws + 173312);                          // 1008 floats
    _Float16* fea2h = (_Float16*)(ws + 177408);                       // N*64 halfs
    _Float16* cat   = (_Float16*)(ws + 177408 + (size_t)N * 64 * 2);  // N*102 halfs

    k0_pack<<<(BTOT + BITOT + 255) / 256, 256, 0, stream>>>(wp, Bp, Bias);
    k1_pointchain<<<(ntiles + 7) / 8, 256, 0, stream>>>(fea, Bp, Bias, fea2h, cat, ntiles);
    k2_neighbors<<<(ntiles + 1) / 2, 64, 0, stream>>>(fea, link, Bp, Bias, fea2h, cat, ntiles);
    k3_head<<<(ntiles * 2 + 3) / 4, 128, 0, stream>>>(cat, cfe, Bp, Bias, (float*)d_out, ntiles);
}
